// CDGRL_88725434400999
// MI455X (gfx1250) — matrix operand order repaired and checked by reading, not compiled, not run on hardware
//
#include <hip/hip_runtime.h>

// ---------------------------------------------------------------------------
// CDGRL pipeline for MI455X (gfx1250, wave32, WMMA bf16->f32, async LDS DMA)
// All GEMMs are C = A[M,K] @ (Bt[N,K])^T : both tiles are straight row copies,
// staged with global_load_async_to_lds_b128 into double-buffered LDS.
// K-step 64: 8 async b128 / wave / stage, 16 WMMAs between barrier pairs.
// ---------------------------------------------------------------------------

typedef __attribute__((ext_vector_type(16))) __bf16 v16bf;
typedef __attribute__((ext_vector_type(8)))  float  v8f;

#define NCLS 21
#define EPSV 1e-8f
#define BM 128
#define BN 128
#define LDSW 36          // dword pitch of an LDS tile row (64 bf16 + 16B pad)

__device__ __forceinline__ v8f wmma_bf16(v16bf a, v16bf b, v8f c) {
  return __builtin_amdgcn_wmma_f32_16x16x32_bf16(false, a, false, b,
                                                 (short)0, c, false, false);
}

union FragU { v16bf v; uint4 q[2]; };

// A-matrix fragment (16x32): lane<16 -> K 0-7 / 16-23 ; lane>=16 -> K 8-15 / 24-31
__device__ __forceinline__ v16bf load_fragA(const unsigned int* row, int h) {
  FragU f;
  f.q[0] = *reinterpret_cast<const uint4*>(row + h * 4);
  f.q[1] = *reinterpret_cast<const uint4*>(row + 8 + h * 4);
  return f.v;
}
// B-matrix fragment (32x16): lane<16 -> K 0-7 / 16-23 ; lane>=16 -> K 8-15 / 24-31
__device__ __forceinline__ v16bf load_fragB(const unsigned int* row, int h) {
  FragU f;
  f.q[0] = *reinterpret_cast<const uint4*>(row + h * 4);
  f.q[1] = *reinterpret_cast<const uint4*>(row + 8 + h * 4);
  return f.v;
}

__device__ __forceinline__ void async_ld16(unsigned int lds_off, const void* g) {
  asm volatile("global_load_async_to_lds_b128 %0, %1, off"
               :: "v"(lds_off), "v"(g) : "memory");
}

// ---------------------------------------------------------------------------
// Shared WMMA main loop (double-buffered, 64-wide K steps, peeled tail).
// ---------------------------------------------------------------------------
__device__ __forceinline__ void mma_mainloop(
    const __bf16* __restrict__ A, const __bf16* __restrict__ Bt,
    int K, int lda, int ldb, int bm, int bn,
    unsigned int* ldsA, unsigned int* ldsB,  // each [2][BM*LDSW]
    v8f acc[2][4]) {
  const int t = threadIdx.x;
  const int lane = t & 31;
  const int wm = (t >> 5) >> 1, wn = (t >> 5) & 1;
  const int h = lane >> 4, l16 = lane & 15;
  const unsigned ldsAbase = (unsigned)(size_t)(void*)ldsA;
  const unsigned ldsBbase = (unsigned)(size_t)(void*)ldsB;
  const int r0 = t >> 3, q8 = (t & 7) * 8;

  auto stage = [&](int buf, int k0) {
#pragma unroll
    for (int it = 0; it < 4; ++it) {
      int r = r0 + it * 32;
      unsigned off = (unsigned)(buf * (BM * LDSW * 4) + r * (LDSW * 4) + q8 * 2);
      async_ld16(ldsAbase + off,
                 (const void*)(A + (size_t)(bm + r) * lda + k0 + q8));
      async_ld16(ldsBbase + off,
                 (const void*)(Bt + (size_t)(bn + r) * ldb + k0 + q8));
    }
  };
  auto compute = [&](int buf) {
    const unsigned int* LA = ldsA + buf * (BM * LDSW);
    const unsigned int* LB = ldsB + buf * (BM * LDSW);
#pragma unroll
    for (int ks = 0; ks < 2; ++ks) {   // two 32-wide K sub-steps
      v16bf af[2], bfr[4];
#pragma unroll
      for (int mi = 0; mi < 2; ++mi)
        af[mi] = load_fragA(&LA[(wm * 32 + mi * 16 + l16) * LDSW + ks * 16], h);
#pragma unroll
      for (int ni = 0; ni < 4; ++ni)
        bfr[ni] = load_fragB(&LB[(wn * 64 + ni * 16 + l16) * LDSW + ks * 16], h);
#pragma unroll
      for (int mi = 0; mi < 2; ++mi)
#pragma unroll
        for (int ni = 0; ni < 4; ++ni)
          acc[mi][ni] = wmma_bf16(af[mi], bfr[ni], acc[mi][ni]);
    }
  };

  stage(0, 0);
  int buf = 0;
  // steady state: branch-free overlap of next-stage DMA with current compute
  for (int k0 = 0; k0 + 64 < K; k0 += 64, buf ^= 1) {
    stage(buf ^ 1, k0 + 64);
    asm volatile("s_wait_asynccnt 8" ::: "memory");  // current stage landed
    __syncthreads();
    compute(buf);
    __syncthreads();
  }
  // tail stage
  asm volatile("s_wait_asynccnt 0" ::: "memory");
  __syncthreads();
  compute(buf);
}

// ---------------------------------------------------------------------------
// Generic GEMM-BT with epilogue: optional per-row scale (dinv), bias, relu,
// optional transposed bf16 store (for producing Bt operands of later GEMMs).
// ---------------------------------------------------------------------------
__global__ __launch_bounds__(256) void gemm_bt_k(
    const __bf16* __restrict__ A, const __bf16* __restrict__ Bt,
    const float* __restrict__ rowScale, const float* __restrict__ bias,
    int relu, int tstore, __bf16* __restrict__ Cb, float* __restrict__ Cf,
    int N, int K, int lda, int ldb, int ldc) {
  __shared__ __align__(16) unsigned int ldsA[2 * BM * LDSW];
  __shared__ __align__(16) unsigned int ldsB[2 * BN * LDSW];
  const int t = threadIdx.x;
  const int lane = t & 31;
  const int wm = (t >> 5) >> 1, wn = (t >> 5) & 1;
  const int bm = blockIdx.y * BM, bn = blockIdx.x * BN;
  const int h = lane >> 4, l16 = lane & 15;

  v8f acc[2][4];
#pragma unroll
  for (int mi = 0; mi < 2; ++mi)
#pragma unroll
    for (int ni = 0; ni < 4; ++ni) acc[mi][ni] = (v8f){0,0,0,0,0,0,0,0};

  mma_mainloop(A, Bt, K, lda, ldb, bm, bn, ldsA, ldsB, acc);

#pragma unroll
  for (int mi = 0; mi < 2; ++mi)
#pragma unroll
    for (int ni = 0; ni < 4; ++ni) {
      int baseRow = bm + wm * 32 + mi * 16 + h * 8;
      int col = bn + wn * 64 + ni * 16 + l16;
#pragma unroll
      for (int e = 0; e < 8; ++e) {
        int i = baseRow + e;
        float vv = acc[mi][ni][e];
        if (rowScale) vv *= rowScale[i];
        if (col < N) {
          if (bias) vv += bias[col];
          if (relu) vv = fmaxf(vv, 0.f);
          if (Cb) {
            if (tstore) Cb[(size_t)col * ldc + i] = (__bf16)vv;
            else        Cb[(size_t)i * ldc + col] = (__bf16)vv;
          }
          if (Cf) Cf[(size_t)i * ldc + col] = vv;
        }
      }
    }
}

// ---------------------------------------------------------------------------
// S builder (symmetric): only upper block-triangle computed; off-diagonal
// tiles are mirrored.  S = mask( |xn@xn^T| * (1-|w_i-w_j|) ) + I.
// rowsum picks up row-sums (shuffle reduce) + mirrored column-sums.
// ---------------------------------------------------------------------------
__global__ __launch_bounds__(256) void sbuild_k(
    const __bf16* __restrict__ Xn, const float* __restrict__ w,
    __bf16* __restrict__ Sb, float* __restrict__ rowsum, int NT, int D) {
  const int bm = blockIdx.y * BM, bn = blockIdx.x * BN;
  if (bn < bm) return;                   // symmetric: upper triangle only
  __shared__ __align__(16) unsigned int ldsA[2 * BM * LDSW];
  __shared__ __align__(16) unsigned int ldsB[2 * BN * LDSW];
  const int t = threadIdx.x;
  const int lane = t & 31;
  const int wm = (t >> 5) >> 1, wn = (t >> 5) & 1;
  const int h = lane >> 4, l16 = lane & 15;
  const bool diag = (bm == bn);

  v8f acc[2][4];
#pragma unroll
  for (int mi = 0; mi < 2; ++mi)
#pragma unroll
    for (int ni = 0; ni < 4; ++ni) acc[mi][ni] = (v8f){0,0,0,0,0,0,0,0};

  mma_mainloop(Xn, Xn, D, D, D, bm, bn, ldsA, ldsB, acc);

  float colsum[4] = {0.f, 0.f, 0.f, 0.f};
#pragma unroll
  for (int mi = 0; mi < 2; ++mi)
#pragma unroll
    for (int ni = 0; ni < 4; ++ni) {
      int baseRow = bm + wm * 32 + mi * 16 + h * 8;
      int col = bn + wn * 64 + ni * 16 + l16;
#pragma unroll
      for (int e = 0; e < 8; ++e) {
        int i = baseRow + e;
        int j = col;
        bool di = i >= 2048, dj = j >= 2048;
        float s;
        if (di == dj) {
          s = (i == j) ? 1.f : 0.f;
        } else {
          float wt = 1.f - fabsf(w[i] - w[j]);
          s = fabsf(acc[mi][ni][e]) * wt;
        }
        Sb[(size_t)i * NT + j] = (__bf16)s;
        if (!diag) {
          Sb[(size_t)j * NT + i] = (__bf16)s;   // mirror
          colsum[ni] += s;                      // contributes to rowsum[j]
        }
        float r = s;                       // reduce cols within 16-lane group
        for (int off = 8; off; off >>= 1) r += __shfl_xor(r, off, 16);
        if (l16 == 0) atomicAdd(&rowsum[i], r);
      }
    }
  if (!diag) {
#pragma unroll
    for (int ni = 0; ni < 4; ++ni)
      atomicAdd(&rowsum[bn + wn * 64 + ni * 16 + l16], colsum[ni]);
  }
}

// ---------------------------------------------------------------------------
// Small helper kernels
// ---------------------------------------------------------------------------
__global__ void prep_x_k(const float* __restrict__ x1,
                         const float* __restrict__ x2,
                         __bf16* __restrict__ xbf, __bf16* __restrict__ xnbf,
                         float* __restrict__ norms, int D) {
  int i = blockIdx.x;
  const float* src = (i < 2048) ? x1 + (size_t)i * D : x2 + (size_t)(i - 2048) * D;
  __shared__ float red[256];
  __shared__ float nrm;
  float local[16];
  float ss = 0.f;
#pragma unroll
  for (int c = 0; c < 16; ++c) {
    float v = src[threadIdx.x + c * 256];
    local[c] = v;
    ss += v * v;
  }
  red[threadIdx.x] = ss;
  __syncthreads();
  for (int s = 128; s > 0; s >>= 1) {
    if (threadIdx.x < s) red[threadIdx.x] += red[threadIdx.x + s];
    __syncthreads();
  }
  if (threadIdx.x == 0) {
    float n = sqrtf(red[0]);
    norms[i] = n;
    nrm = fmaxf(n, EPSV);
  }
  __syncthreads();
  float inv = 1.f / nrm;
#pragma unroll
  for (int c = 0; c < 16; ++c) {
    int k = threadIdx.x + c * 256;
    xbf[(size_t)i * D + k] = (__bf16)local[c];
    xnbf[(size_t)i * D + k] = (__bf16)(local[c] * inv);
  }
}

// transpose-convert: src[K][N] f32  ->  dst[N][K] bf16
__global__ void f2bf_t_k(const float* __restrict__ src, __bf16* __restrict__ dst,
                         int K, int N) {
  int i = blockIdx.x * blockDim.x + threadIdx.x;
  if (i < K * N) {
    int r = i / N, c = i % N;
    dst[(size_t)c * K + r] = (__bf16)src[i];
  }
}

// fw3[256][21] f32 -> fw3t[128][256] bf16 (transposed, zero-padded rows)
__global__ void padw3t_k(const float* __restrict__ fw3, __bf16* __restrict__ dst) {
  int i = blockIdx.x * blockDim.x + threadIdx.x;  // 128*256
  int n = i >> 8, k = i & 255;
  dst[i] = (n < NCLS) ? (__bf16)fw3[k * NCLS + n] : (__bf16)0.f;
}

__global__ void padb3_k(const float* __restrict__ fb3, float* __restrict__ d) {
  int i = threadIdx.x;
  if (i < 32) d[i] = (i < NCLS) ? fb3[i] : 0.f;
}

__global__ void count_k(const int* __restrict__ l1, const int* __restrict__ l2,
                        int* __restrict__ counts) {
  int i = blockIdx.x * blockDim.x + threadIdx.x;
  if (i >= 4096) return;
  int dom = i >= 2048;
  int lab = dom ? l2[i - 2048] : l1[i];
  atomicAdd(&counts[dom * NCLS + lab], 1);
}

__global__ void cent_acc_k(const float* __restrict__ x1, const float* __restrict__ x2,
                           const int* __restrict__ l1, const int* __restrict__ l2,
                           float* __restrict__ cent, int D) {
  int i = blockIdx.x;
  int dom = i >= 2048;
  int li = dom ? i - 2048 : i;
  const float* src = dom ? x2 + (size_t)li * D : x1 + (size_t)li * D;
  int lab = dom ? l2[li] : l1[li];
  float* c = cent + ((size_t)dom * NCLS + lab) * D;
#pragma unroll
  for (int ck = 0; ck < 16; ++ck) {
    int k = threadIdx.x + ck * 256;
    atomicAdd(&c[k], src[k]);
  }
}

__global__ void cent_fin_k(float* __restrict__ cent, const int* __restrict__ counts,
                           float* __restrict__ ctnorm, int D) {
  int idx = blockIdx.x;  // 0..41
  float cnt = fmaxf((float)counts[idx], 1.f);
  __shared__ float red[256];
  float ss = 0.f;
  for (int k = threadIdx.x; k < D; k += 256) {
    float v = floorf(cent[(size_t)idx * D + k] / cnt);
    cent[(size_t)idx * D + k] = v;
    ss += v * v;
  }
  red[threadIdx.x] = ss;
  __syncthreads();
  for (int s = 128; s > 0; s >>= 1) {
    if (threadIdx.x < s) red[threadIdx.x] += red[threadIdx.x + s];
    __syncthreads();
  }
  if (threadIdx.x == 0) ctnorm[idx] = sqrtf(red[0]);
}

__global__ void simi_k(const float* __restrict__ x1, const float* __restrict__ x2,
                       const int* __restrict__ l1, const int* __restrict__ l2,
                       const float* __restrict__ cent, const float* __restrict__ ctnorm,
                       const float* __restrict__ norms, float* __restrict__ simi,
                       unsigned int* __restrict__ clsmax, int D) {
  int i = blockIdx.x;
  int dom = i >= 2048;
  int li = dom ? i - 2048 : i;
  int lab = dom ? l2[li] : l1[li];
  const float* src = dom ? x2 + (size_t)li * D : x1 + (size_t)li * D;
  const float* c = cent + ((size_t)dom * NCLS + lab) * D;
  __shared__ float red[256];
  float dt = 0.f;
  for (int k = threadIdx.x; k < D; k += 256) dt += src[k] * c[k];
  red[threadIdx.x] = dt;
  __syncthreads();
  for (int s = 128; s > 0; s >>= 1) {
    if (threadIdx.x < s) red[threadIdx.x] += red[threadIdx.x + s];
    __syncthreads();
  }
  if (threadIdx.x == 0) {
    float den = fmaxf(norms[i], EPSV) * fmaxf(ctnorm[dom * NCLS + lab], EPSV);
    float s = fabsf(red[0] / den);
    simi[i] = s;
    atomicMax(&clsmax[dom * NCLS + lab], __float_as_uint(s));  // s >= 0
  }
}

__global__ void wfin_k(const int* __restrict__ l1, const int* __restrict__ l2,
                       const float* __restrict__ simi,
                       const unsigned int* __restrict__ clsmax,
                       float* __restrict__ w) {
  int i = blockIdx.x * blockDim.x + threadIdx.x;
  if (i >= 4096) return;
  int dom = i >= 2048;
  int lab = dom ? l2[i - 2048] : l1[i];
  float mx = __uint_as_float(clsmax[dom * NCLS + lab]);
  if (mx == 0.f) mx = 1.f;
  w[i] = simi[i] / mx;
}

__global__ void dinv_k(const float* __restrict__ rowsum, float* __restrict__ dinv) {
  int i = blockIdx.x * blockDim.x + threadIdx.x;
  if (i < 4096) dinv[i] = rsqrtf(rowsum[i]);
}

__global__ void loss_k(const float* __restrict__ logits, const int* __restrict__ l1,
                       const int* __restrict__ l2, float* __restrict__ out) {
  int i = blockIdx.x * blockDim.x + threadIdx.x;
  if (i >= 4096) return;
  const float* l = logits + (size_t)i * 32;
  float m = -1e30f;
  for (int c = 0; c < NCLS; ++c) m = fmaxf(m, l[c]);
  float se = 0.f;
  for (int c = 0; c < NCLS; ++c) se += __expf(l[c] - m);
  int lab = (i < 2048) ? l1[i] : l2[i - 2048];
  float lp = l[lab] - m - __logf(se);
  atomicAdd(out, -lp * (1.f / 4096.f));
}

// ---------------------------------------------------------------------------
// Host-side orchestration
// ---------------------------------------------------------------------------
extern "C" void kernel_launch(void* const* d_in, const int* in_sizes, int n_in,
                              void* d_out, int out_size, void* d_ws, size_t ws_size,
                              hipStream_t stream) {
  (void)in_sizes; (void)n_in; (void)out_size; (void)ws_size;
  const int NT = 4096, D = 4096;

  const float* x1 = (const float*)d_in[0];
  const float* x2 = (const float*)d_in[1];
  const int* l1 = (const int*)d_in[2];
  const int* l2 = (const int*)d_in[3];
  const float* W1 = (const float*)d_in[4];
  const float* b1 = (const float*)d_in[5];
  const float* W2 = (const float*)d_in[6];
  const float* b2 = (const float*)d_in[7];
  const float* fw1 = (const float*)d_in[8];
  const float* fb1 = (const float*)d_in[9];
  const float* fw2 = (const float*)d_in[10];
  const float* fb2 = (const float*)d_in[11];
  const float* fw3 = (const float*)d_in[12];
  const float* fb3 = (const float*)d_in[13];
  float* out = (float*)d_out;

  char* p = (char*)d_ws;
  auto alloc = [&](size_t bytes) -> char* {
    char* r = p;
    p += (bytes + 255) & ~(size_t)255;
    return r;
  };
  __bf16* xbf   = (__bf16*)alloc((size_t)NT * D * 2);
  __bf16* xnbf  = (__bf16*)alloc((size_t)NT * D * 2);
  __bf16* W1t   = (__bf16*)alloc((size_t)2048 * 4096 * 2);
  __bf16* W2t   = (__bf16*)alloc((size_t)1024 * 2048 * 2);
  __bf16* fw1t  = (__bf16*)alloc((size_t)512 * 1024 * 2);
  __bf16* fw2t  = (__bf16*)alloc((size_t)256 * 512 * 2);
  __bf16* fw3t  = (__bf16*)alloc((size_t)128 * 256 * 2);  // padded to 128 rows
  float*  fb3p  = (float*)alloc(32 * 4);
  __bf16* Sbf   = (__bf16*)alloc((size_t)NT * NT * 2);
  __bf16* T1t   = (__bf16*)alloc((size_t)2048 * NT * 2);  // (x@W1*dinv)^T
  __bf16* H1    = (__bf16*)alloc((size_t)NT * 2048 * 2);
  __bf16* T2t   = (__bf16*)alloc((size_t)1024 * NT * 2);  // (H1@W2*dinv)^T
  __bf16* H2    = (__bf16*)alloc((size_t)NT * 1024 * 2);
  __bf16* H3    = (__bf16*)alloc((size_t)NT * 512 * 2);
  __bf16* H4    = (__bf16*)alloc((size_t)NT * 256 * 2);
  float*  logits = (float*)alloc((size_t)NT * 32 * 4);
  float*  norms = (float*)alloc(NT * 4);
  float*  w     = (float*)alloc(NT * 4);
  float*  simi  = (float*)alloc(NT * 4);
  float*  dinv  = (float*)alloc(NT * 4);
  float*  rowsum = (float*)alloc(NT * 4);
  float*  cent  = (float*)alloc((size_t)2 * NCLS * D * 4);
  float*  ctnorm = (float*)alloc(2 * NCLS * 4);
  int*    counts = (int*)alloc(2 * NCLS * 4);
  unsigned int* clsmax = (unsigned int*)alloc(2 * NCLS * 4);

  // zero accumulators (re-done every call; harness poisons ws only once)
  hipMemsetAsync(cent, 0, (size_t)2 * NCLS * D * 4, stream);
  hipMemsetAsync(counts, 0, 2 * NCLS * 4, stream);
  hipMemsetAsync(clsmax, 0, 2 * NCLS * 4, stream);
  hipMemsetAsync(rowsum, 0, NT * 4, stream);
  hipMemsetAsync(out, 0, 4, stream);

  // ---- weights (_weighted) + conversions ----
  prep_x_k<<<NT, 256, 0, stream>>>(x1, x2, xbf, xnbf, norms, D);
  f2bf_t_k<<<(4096 * 2048 + 255) / 256, 256, 0, stream>>>(W1, W1t, 4096, 2048);
  f2bf_t_k<<<(2048 * 1024 + 255) / 256, 256, 0, stream>>>(W2, W2t, 2048, 1024);
  f2bf_t_k<<<(1024 * 512 + 255) / 256, 256, 0, stream>>>(fw1, fw1t, 1024, 512);
  f2bf_t_k<<<(512 * 256 + 255) / 256, 256, 0, stream>>>(fw2, fw2t, 512, 256);
  padw3t_k<<<128, 256, 0, stream>>>(fw3, fw3t);
  padb3_k<<<1, 32, 0, stream>>>(fb3, fb3p);
  count_k<<<16, 256, 0, stream>>>(l1, l2, counts);
  cent_acc_k<<<NT, 256, 0, stream>>>(x1, x2, l1, l2, cent, D);
  cent_fin_k<<<2 * NCLS, 256, 0, stream>>>(cent, counts, ctnorm, D);
  simi_k<<<NT, 256, 0, stream>>>(x1, x2, l1, l2, cent, ctnorm, norms, simi, clsmax, D);
  wfin_k<<<16, 256, 0, stream>>>(l1, l2, simi, clsmax, w);

  // ---- graph matrix S (symmetric; upper block triangle only) ----
  sbuild_k<<<dim3(NT / BN, NT / BM), 256, 0, stream>>>(xnbf, w, Sbf, rowsum, NT, D);
  dinv_k<<<16, 256, 0, stream>>>(rowsum, dinv);

  // ---- GCN with D^{-1/2} folded into row scales ----
  // T1t = (dinv .* (x @ W1))^T
  gemm_bt_k<<<dim3(16, 32), 256, 0, stream>>>(
      xbf, W1t, dinv, nullptr, 0, 1, T1t, nullptr, 2048, 4096, 4096, 4096, 4096);
  // H1 = relu(dinv .* (S @ T1) + b1)
  gemm_bt_k<<<dim3(16, 32), 256, 0, stream>>>(
      Sbf, T1t, dinv, b1, 1, 0, H1, nullptr, 2048, 4096, 4096, 4096, 2048);
  // T2t = (dinv .* (H1 @ W2))^T
  gemm_bt_k<<<dim3(8, 32), 256, 0, stream>>>(
      H1, W2t, dinv, nullptr, 0, 1, T2t, nullptr, 1024, 2048, 2048, 2048, 4096);
  // H2 = dinv .* (S @ T2) + b2
  gemm_bt_k<<<dim3(8, 32), 256, 0, stream>>>(
      Sbf, T2t, dinv, b2, 0, 0, H2, nullptr, 1024, 4096, 4096, 4096, 1024);

  // ---- classifier ----
  gemm_bt_k<<<dim3(4, 32), 256, 0, stream>>>(
      H2, fw1t, nullptr, fb1, 1, 0, H3, nullptr, 512, 1024, 1024, 1024, 512);
  gemm_bt_k<<<dim3(2, 32), 256, 0, stream>>>(
      H3, fw2t, nullptr, fb2, 1, 0, H4, nullptr, 256, 512, 512, 512, 256);
  gemm_bt_k<<<dim3(1, 32), 256, 0, stream>>>(
      H4, fw3t, nullptr, fb3p, 0, 0, nullptr, logits, 32, 256, 256, 256, 32);

  // ---- NLL of log_softmax ----
  loss_k<<<16, 256, 0, stream>>>(logits, l1, l2, out);
}
